// GGM_48524540510637
// MI455X (gfx1250) — compile-verified, matrix operand order checked by reading, not run on hardware
//
#include <hip/hip_runtime.h>
#include <cstddef>

#define DD 128
#define NT 256

typedef __bf16 bf16;
typedef bf16  v16bf __attribute__((ext_vector_type(16)));
typedef bf16  v8bf  __attribute__((ext_vector_type(8)));
typedef float v8f   __attribute__((ext_vector_type(8)));

// ---------------------------------------------------------------------------
// WMMA helpers (v_wmma_f32_16x16x32_bf16, wave32)
// A fragment (16xK row-major, per ISA 7.12.2): lane l<16 -> row l, K {0..7,16..23}
//                                              lane l>=16 -> row l-16, K {8..15,24..31}
// B fragment loaded from row-major W (N x K): symmetric, row index = output col.
// ---------------------------------------------------------------------------
__device__ __forceinline__ v16bf ldfrag(const bf16* base, int ld, int lane) {
  int row  = lane & 15;
  int koff = (lane & 16) ? 8 : 0;
  const bf16* p = base + (size_t)row * ld + koff;
  v8bf lo = *reinterpret_cast<const v8bf*>(p);
  v8bf hi = *reinterpret_cast<const v8bf*>(p + 16);
  v16bf f;
#pragma unroll
  for (int i = 0; i < 8; ++i) { f[i] = lo[i]; f[8 + i] = hi[i]; }
  return f;
}

__device__ __forceinline__ v8f wmma_bf16(v16bf a, v16bf b, v8f c) {
  return __builtin_amdgcn_wmma_f32_16x16x32_bf16(false, a, false, b, (short)0, c,
                                                 false, false);
}

__device__ __forceinline__ float sigmoidf_(float x) {
  return 1.0f / (1.0f + __expf(-x));
}

// ---------------------------------------------------------------------------
// Prep kernels
// ---------------------------------------------------------------------------
__global__ void conv_bf16_k(bf16* dst, const float* src, int n) {
  int i = blockIdx.x * blockDim.x + threadIdx.x;
  if (i < n) dst[i] = (bf16)src[i];
}

// encU_W: (3, 128, 388) -> packed bf16 (3, 3*128, 128): rows 0..127=Wa,128..255=Wb,256..383=Wc
__global__ void pack_encU_k(bf16* dst, const float* src) {
  int i = blockIdx.x * blockDim.x + threadIdx.x;   // 3*3*128*128
  if (i >= 3 * 3 * 128 * 128) return;
  int k   = i & 127;
  int d   = (i >> 7) & 127;
  int seg = (i >> 14) % 3;
  int l   = i / (3 * 128 * 128);
  dst[i] = (bf16)src[((size_t)l * 128 + d) * 388 + seg * 128 + k];
}

__global__ void zero_f32_k(float* p, int n) {
  int i = blockIdx.x * blockDim.x + threadIdx.x;
  if (i < n) p[i] = 0.0f;
}

// h = x_atom @ W_node.T  (FA=16), write f32 + bf16
__global__ void init_nodes_k(const float* x, const float* Wn, float* hf, bf16* hb, int n) {
  int i = blockIdx.x * blockDim.x + threadIdx.x;   // n*128
  if (i >= n * DD) return;
  int v = i >> 7, d = i & 127;
  float s = 0.0f;
#pragma unroll
  for (int f = 0; f < 16; ++f) s += x[(size_t)v * 16 + f] * Wn[d * 16 + f];
  hf[i] = s;
  hb[i] = (bf16)s;
}

// e = e_bond @ W_edge.T  (FB=10), write bf16 only (used solely as WMMA A input)
__global__ void init_edges_k(const float* eb, const float* We, bf16* ebf, int n) {
  int i = blockIdx.x * blockDim.x + threadIdx.x;   // n*128
  if (i >= n * DD) return;
  int ed = i >> 7, d = i & 127;
  float s = 0.0f;
#pragma unroll
  for (int f = 0; f < 10; ++f) s += eb[(size_t)ed * 10 + f] * We[d * 10 + f];
  ebf[i] = (bf16)s;
}

// cvec = condition @ Wd.T + encU_b[l]   (Wd = encU_W[l][:, 384:388])
__global__ void cvec_k(const float* cond, const float* encW, const float* encb, float* cvec) {
  int d = threadIdx.x;   // 128
  float s = encb[d];
#pragma unroll
  for (int c = 0; c < 4; ++c) s += cond[c] * encW[(size_t)d * 388 + 384 + c];
  cvec[d] = s;
}

// ---------------------------------------------------------------------------
// hab[n, 0:128]=h@Wa.T, hab[n,128:256]=h@Wb.T.  Block: 16 nodes, 8 waves x 2 n-tiles.
// ---------------------------------------------------------------------------
__global__ void gemm_hab_k(const bf16* __restrict__ hbf, const bf16* __restrict__ Wab,
                           float* __restrict__ hab) {
  int lane = threadIdx.x & 31;
  int wv   = threadIdx.x >> 5;
  size_t node0 = (size_t)blockIdx.x * 16;
  const bf16* A = hbf + node0 * DD;
  for (int nt = wv; nt < 16; nt += 8) {
    v8f c = {};
#pragma unroll
    for (int kc = 0; kc < 4; ++kc) {
      v16bf a = ldfrag(A + kc * 32, DD, lane);
      v16bf b = ldfrag(Wab + (size_t)(nt * 16) * DD + kc * 32, DD, lane);
      c = wmma_bf16(a, b, c);
    }
    int nloc = lane & 15;
    int mb   = (lane & 16) ? 8 : 0;
#pragma unroll
    for (int r = 0; r < 8; ++r)
      hab[(node0 + mb + r) * 256 + nt * 16 + nloc] = c[r];
  }
}

// ---------------------------------------------------------------------------
// Fused ec = e@Wc.T tile + message assembly + segment-sum scatter:
//   agg[dst] += ec + ha[src] + hb[dst] + cvec
// Block: 16 edges, 8 waves x 1 n-tile (cols 0..127).
// ---------------------------------------------------------------------------
__global__ void edge_msg_scatter_k(const bf16* __restrict__ ebf, const bf16* __restrict__ Wc,
                                   const float* __restrict__ hab, const float* __restrict__ cvec,
                                   const int* __restrict__ src, const int* __restrict__ dst,
                                   float* agg) {
  int lane = threadIdx.x & 31;
  int wv   = threadIdx.x >> 5;
  size_t e0 = (size_t)blockIdx.x * 16;
  const bf16* A = ebf + e0 * DD;
  v8f c = {};
#pragma unroll
  for (int kc = 0; kc < 4; ++kc) {
    v16bf a = ldfrag(A + kc * 32, DD, lane);
    v16bf b = ldfrag(Wc + (size_t)(wv * 16) * DD + kc * 32, DD, lane);
    c = wmma_bf16(a, b, c);
  }
  int nloc = lane & 15;
  int mb   = (lane & 16) ? 8 : 0;
  int col  = wv * 16 + nloc;
  float cv = cvec[col];
#pragma unroll
  for (int r = 0; r < 8; ++r) {
    size_t ed = e0 + mb + r;
    int s  = src[ed];
    int dd = dst[ed];
    float val = c[r] + hab[(size_t)s * 256 + col] + hab[(size_t)dd * 256 + 128 + col] + cv;
    atomicAdd(&agg[(size_t)dd * DD + col], val);
  }
}

// ---------------------------------------------------------------------------
// Fused GRU: gi = agg@Wih.T + bih, gh = h@Whh.T + bhh, then gate math -> new h.
// Block: 16 nodes, 8 waves over 48 output tiles (24 gi + 24 gh). In-place on h.
// ---------------------------------------------------------------------------
__global__ void gru_fused_k(const float* agg, const float* hf32, const bf16* hbf,
                            const bf16* __restrict__ Wih, const bf16* __restrict__ Whh,
                            const float* __restrict__ bih, const float* __restrict__ bhh,
                            float* hf_out, bf16* hb_out) {
  __shared__ bf16  sA[16 * DD];
  __shared__ bf16  sH[16 * DD];
  __shared__ float sGi[16 * 384];
  __shared__ float sGh[16 * 384];
  size_t node0 = (size_t)blockIdx.x * 16;
  for (int i = threadIdx.x; i < 16 * DD; i += blockDim.x) {
    sA[i] = (bf16)agg[node0 * DD + i];
    sH[i] = hbf[node0 * DD + i];
  }
  __syncthreads();
  int lane = threadIdx.x & 31;
  int wv   = threadIdx.x >> 5;
  for (int t = wv; t < 48; t += 8) {
    bool is_gh = (t >= 24);
    int ct = is_gh ? t - 24 : t;
    const bf16* A = is_gh ? sH : sA;
    const bf16* W = is_gh ? Whh : Wih;
    float* out    = is_gh ? sGh : sGi;
    v8f c = {};
#pragma unroll
    for (int kc = 0; kc < 4; ++kc) {
      v16bf a = ldfrag(A + kc * 32, DD, lane);
      v16bf b = ldfrag(W + (size_t)(ct * 16) * DD + kc * 32, DD, lane);
      c = wmma_bf16(a, b, c);
    }
    int nloc = lane & 15;
    int mb   = (lane & 16) ? 8 : 0;
#pragma unroll
    for (int r = 0; r < 8; ++r)
      out[(mb + r) * 384 + ct * 16 + nloc] = c[r];
  }
  __syncthreads();
  for (int i = threadIdx.x; i < 16 * DD; i += blockDim.x) {
    int n = i >> 7, d = i & 127;
    float gir = sGi[n * 384 + d]       + bih[d];
    float giz = sGi[n * 384 + 128 + d] + bih[128 + d];
    float gin = sGi[n * 384 + 256 + d] + bih[256 + d];
    float ghr = sGh[n * 384 + d]       + bhh[d];
    float ghz = sGh[n * 384 + 128 + d] + bhh[128 + d];
    float ghn = sGh[n * 384 + 256 + d] + bhh[256 + d];
    float r  = sigmoidf_(gir + ghr);
    float z  = sigmoidf_(giz + ghz);
    float nn = tanhf(gin + r * ghn);
    float hnew = (1.0f - z) * nn + z * hf32[node0 * DD + i];
    hf_out[node0 * DD + i] = hnew;
    hb_out[node0 * DD + i] = (bf16)hnew;
  }
}

// ---------------------------------------------------------------------------
// Readout: hv = tanh(h@evW.T+b) * sigmoid(h@evcW.T+b); colsum += per-block sums.
// Block: 16 nodes, 8 waves x 2 n-tiles (evW cols 0..127, evcW cols 128..255).
// ---------------------------------------------------------------------------
__global__ void readout_k(const bf16* __restrict__ hbf,
                          const bf16* __restrict__ evW, const bf16* __restrict__ evcW,
                          const float* __restrict__ evb, const float* __restrict__ evcb,
                          float* colsum) {
  __shared__ float sC[16 * 256];
  int lane = threadIdx.x & 31;
  int wv   = threadIdx.x >> 5;
  size_t node0 = (size_t)blockIdx.x * 16;
  const bf16* A = hbf + node0 * DD;
  for (int nt = wv; nt < 16; nt += 8) {
    const bf16* W = (nt >= 8) ? evcW : evW;
    int nb = (nt & 7) * 16;
    v8f c = {};
#pragma unroll
    for (int kc = 0; kc < 4; ++kc) {
      v16bf a = ldfrag(A + kc * 32, DD, lane);
      v16bf b = ldfrag(W + (size_t)nb * DD + kc * 32, DD, lane);
      c = wmma_bf16(a, b, c);
    }
    int nloc = lane & 15;
    int mb   = (lane & 16) ? 8 : 0;
#pragma unroll
    for (int r = 0; r < 8; ++r)
      sC[(mb + r) * 256 + nt * 16 + nloc] = c[r];
  }
  __syncthreads();
  if (threadIdx.x < DD) {
    int d = threadIdx.x;
    float s = 0.0f;
#pragma unroll
    for (int n = 0; n < 16; ++n) {
      float a = sC[n * 256 + d] + evb[d];
      float g = sC[n * 256 + 128 + d] + evcb[d];
      s += tanhf(a) * sigmoidf_(g);
    }
    atomicAdd(&colsum[d], s);
  }
}

// ---------------------------------------------------------------------------
// Finalize: ev = colsum/N; mu, logvar; z = mu + eps*exp(0.5*logvar); concat cond.
// ---------------------------------------------------------------------------
__global__ void finalize_k(const float* colsum, const float* meanW, const float* meanb,
                           const float* logW, const float* logb, const float* eps,
                           const float* cond, float* out, float inv_n) {
  __shared__ float ev[DD];
  int d = threadIdx.x;   // 128 threads
  ev[d] = colsum[d] * inv_n;
  __syncthreads();
  float mu = meanb[d], lv = logb[d];
#pragma unroll 4
  for (int k = 0; k < DD; ++k) {
    mu += ev[k] * meanW[d * DD + k];
    lv += ev[k] * logW[d * DD + k];
  }
  out[d] = mu + eps[d] * __expf(0.5f * lv);
  if (d < 4) out[DD + d] = cond[d];
}

// ---------------------------------------------------------------------------
extern "C" void kernel_launch(void* const* d_in, const int* in_sizes, int n_in,
                              void* d_out, int out_size, void* d_ws, size_t ws_size,
                              hipStream_t stream) {
  const float* x_atom    = (const float*)d_in[0];
  const float* e_bond    = (const float*)d_in[1];
  const int*   src       = (const int*)d_in[2];
  const int*   dst       = (const int*)d_in[3];
  const float* condition = (const float*)d_in[4];
  const float* eps       = (const float*)d_in[5];
  const float* W_node    = (const float*)d_in[6];
  const float* W_edge    = (const float*)d_in[7];
  const float* encU_W    = (const float*)d_in[8];
  const float* encU_b    = (const float*)d_in[9];
  const float* gru_Wih   = (const float*)d_in[10];
  const float* gru_Whh   = (const float*)d_in[11];
  const float* gru_bih   = (const float*)d_in[12];
  const float* gru_bhh   = (const float*)d_in[13];
  const float* ev_W      = (const float*)d_in[14];
  const float* ev_b      = (const float*)d_in[15];
  const float* evc_W     = (const float*)d_in[16];
  const float* evc_b     = (const float*)d_in[17];
  const float* mean_W    = (const float*)d_in[18];
  const float* mean_b    = (const float*)d_in[19];
  const float* logvar_W  = (const float*)d_in[20];
  const float* logvar_b  = (const float*)d_in[21];

  const int n_nodes = in_sizes[0] / 16;   // FA = 16
  const int n_edges = in_sizes[2];

  // workspace layout (all chunks 256B-aligned)
  char* ws = (char*)d_ws;
  float* h_f32  = (float*)ws;  ws += (size_t)n_nodes * DD * 4;
  float* agg    = (float*)ws;  ws += (size_t)n_nodes * DD * 4;
  float* hab    = (float*)ws;  ws += (size_t)n_nodes * 256 * 4;
  bf16*  h_bf   = (bf16*)ws;   ws += (size_t)n_nodes * DD * 2;
  bf16*  e_bf   = (bf16*)ws;   ws += (size_t)n_edges * DD * 2;
  bf16*  encABC = (bf16*)ws;   ws += (size_t)3 * 3 * 128 * 128 * 2;
  bf16*  wih_bf = (bf16*)ws;   ws += (size_t)3 * 384 * 128 * 2;
  bf16*  whh_bf = (bf16*)ws;   ws += (size_t)3 * 384 * 128 * 2;
  bf16*  evw_bf = (bf16*)ws;   ws += (size_t)128 * 128 * 2;
  bf16*  evcw_bf= (bf16*)ws;   ws += (size_t)128 * 128 * 2;
  float* cvec   = (float*)ws;  ws += 512;
  float* colsum = (float*)ws;  ws += 512;

  // ---- weight prep (bf16, padded ld=128 layouts) ----
  pack_encU_k<<<(3 * 3 * 128 * 128 + NT - 1) / NT, NT, 0, stream>>>(encABC, encU_W);
  conv_bf16_k<<<(3 * 384 * 128 + NT - 1) / NT, NT, 0, stream>>>(wih_bf, gru_Wih, 3 * 384 * 128);
  conv_bf16_k<<<(3 * 384 * 128 + NT - 1) / NT, NT, 0, stream>>>(whh_bf, gru_Whh, 3 * 384 * 128);
  conv_bf16_k<<<(128 * 128 + NT - 1) / NT, NT, 0, stream>>>(evw_bf, ev_W, 128 * 128);
  conv_bf16_k<<<(128 * 128 + NT - 1) / NT, NT, 0, stream>>>(evcw_bf, evc_W, 128 * 128);

  // ---- embeddings ----
  init_nodes_k<<<((size_t)n_nodes * DD + NT - 1) / NT, NT, 0, stream>>>(x_atom, W_node, h_f32, h_bf, n_nodes);
  init_edges_k<<<((size_t)n_edges * DD + NT - 1) / NT, NT, 0, stream>>>(e_bond, W_edge, e_bf, n_edges);

  // ---- message-passing layers ----
  for (int l = 0; l < 3; ++l) {
    const bf16* Wab = encABC + (size_t)l * 3 * 128 * 128;            // rows 0..255
    const bf16* Wc  = Wab + (size_t)256 * 128;                       // rows 256..383
    cvec_k<<<1, 128, 0, stream>>>(condition, encU_W + (size_t)l * 128 * 388,
                                  encU_b + (size_t)l * 128, cvec);
    zero_f32_k<<<((size_t)n_nodes * DD + NT - 1) / NT, NT, 0, stream>>>(agg, n_nodes * DD);
    gemm_hab_k<<<n_nodes / 16, NT, 0, stream>>>(h_bf, Wab, hab);
    edge_msg_scatter_k<<<n_edges / 16, NT, 0, stream>>>(e_bf, Wc, hab, cvec, src, dst, agg);
    gru_fused_k<<<n_nodes / 16, NT, 0, stream>>>(agg, h_f32, h_bf,
                                                 wih_bf + (size_t)l * 384 * 128,
                                                 whh_bf + (size_t)l * 384 * 128,
                                                 gru_bih + (size_t)l * 384,
                                                 gru_bhh + (size_t)l * 384,
                                                 h_f32, h_bf);
  }

  // ---- readout ----
  zero_f32_k<<<1, 128, 0, stream>>>(colsum, 128);
  readout_k<<<n_nodes / 16, NT, 0, stream>>>(h_bf, evw_bf, evcw_bf, ev_b, evc_b, colsum);
  finalize_k<<<1, 128, 0, stream>>>(colsum, mean_W, mean_b, logvar_W, logvar_b, eps,
                                    condition, (float*)d_out, 1.0f / (float)n_nodes);
}
